// DynamicQueryAttention_83597243449878
// MI455X (gfx1250) — matrix operand order repaired and checked by reading, not compiled, not run on hardware
//
#include <hip/hip_runtime.h>

// ---------------------------------------------------------------------------
// CDNA5 (gfx1250): bf16 WMMA GEMMs (f32 accum), double-buffered LDS software
// pipeline, v_perm_b32-packed f32->bf16 conversion; 4-step recurrence
// collapsed to per-row scalar math.
// ---------------------------------------------------------------------------

typedef __attribute__((ext_vector_type(16))) __bf16 bf16x16;
typedef __attribute__((ext_vector_type(8)))  float v8f;

#define M_BLK 128
#define N_BLK 64
#define K_BLK 32
#define A_PITCH 40   // bf16 elems per LDS row (80B rows, 16B aligned)
#define B_PITCH 40
#define N_STEPS 4

// Packed f32x2 -> bf16x2. Preference order:
//  1) v_cvt_pk_bf16_f32 (1 op)  2) bias + v_perm_b32 (3 ops)  3) shift/or
__device__ __forceinline__ unsigned pk_bf16(float lo, float hi) {
#if __has_builtin(__builtin_amdgcn_cvt_pk_bf16_f32)
  typedef __attribute__((ext_vector_type(2))) __bf16 bf16x2;
  union { bf16x2 v; unsigned u; } cv;
  cv.v = __builtin_amdgcn_cvt_pk_bf16_f32(lo, hi);
  return cv.u;
#else
  unsigned ul = __float_as_uint(lo) + 0x8000u;   // round-half-up to bf16
  unsigned uh = __float_as_uint(hi) + 0x8000u;
#if __has_builtin(__builtin_amdgcn_perm)
  // result bytes: {hi[3],hi[2],lo[3],lo[2]}  (perm concat = {src0:hi, src1:lo})
  return __builtin_amdgcn_perm(uh, ul, 0x07060302u);
#else
  return (ul >> 16) | (uh & 0xFFFF0000u);
#endif
#endif
}

union FragU { uint4 u[2]; bf16x16 v; };

// ---------------------------------------------------------------------------
// Batched GEMM+bias: out[z] = X[z] @ W[z] + b[z]
// z=0: k_v ; z=1: k_t ; z=2: P_v = e_v@Wq+bq ; z=3: P_t = e_t@Wq+bq
// ---------------------------------------------------------------------------
__global__ __launch_bounds__(256) void gemm_bias_bf16(
    const float* __restrict__ x_v, const float* __restrict__ x_t,
    const float* __restrict__ e_v, const float* __restrict__ e_t,
    const float* __restrict__ Wk_v, const float* __restrict__ Wk_t,
    const float* __restrict__ Wq,
    const float* __restrict__ bk_v, const float* __restrict__ bk_t,
    const float* __restrict__ bq,
    float* __restrict__ ws, int B, int D, int H) {
  __shared__ __align__(16) unsigned short Alds[2][M_BLK * A_PITCH];
  __shared__ __align__(16) unsigned short Btlds[2][N_BLK * B_PITCH];

  const float* X; const float* W; const float* bias; float* out;
  const size_t BH = (size_t)B * H;
  switch (blockIdx.z) {
    case 0:  X = x_v; W = Wk_v; bias = bk_v; out = ws;           break;
    case 1:  X = x_t; W = Wk_t; bias = bk_t; out = ws + BH;      break;
    case 2:  X = e_v; W = Wq;   bias = bq;   out = ws + 2 * BH;  break;
    default: X = e_t; W = Wq;   bias = bq;   out = ws + 3 * BH;  break;
  }

  const int tid  = threadIdx.x;
  const int lane = tid & 31;
  const int wave = tid >> 5;
  const int l15  = lane & 15;
  const int hiA  = (lane >> 4) << 3;   // 0 or 8
  const int hiB  = (lane >> 4) << 4;   // 0 or 16
  const int wm   = wave & 3;
  const int wn   = wave >> 2;
  const int m_base = blockIdx.y * M_BLK;
  const int n_base = blockIdx.x * N_BLK;

  const float* Abase = X + (size_t)m_base * D;
  const float* Wbase = W + n_base;

  const int kp = tid >> 4;             // 0..15 : K-pair for W tile
  const int nq = (tid & 15) << 2;      // 0..60 : N quad for W tile

  float4 ra[4];                        // A tile in flight
  float4 rw0, rw1;                     // W tile in flight

  auto GLOAD = [&](int k0) {
#pragma unroll
    for (int t = 0; t < 4; ++t) {
      int i4  = tid + t * 256;
      int row = i4 >> 3;
      int kq  = (i4 & 7) << 2;
      ra[t] = *(const float4*)(Abase + (size_t)row * D + k0 + kq);
    }
    rw0 = *(const float4*)(Wbase + (size_t)(k0 + 2 * kp) * H + nq);
    rw1 = *(const float4*)(Wbase + (size_t)(k0 + 2 * kp + 1) * H + nq);
  };
  auto SSTORE = [&](int buf) {
#pragma unroll
    for (int t = 0; t < 4; ++t) {
      int i4  = tid + t * 256;
      int row = i4 >> 3;
      int kq  = (i4 & 7) << 2;
      uint2 p;
      p.x = pk_bf16(ra[t].x, ra[t].y);
      p.y = pk_bf16(ra[t].z, ra[t].w);
      *(uint2*)&Alds[buf][row * A_PITCH + kq] = p;
    }
    *(unsigned*)&Btlds[buf][(nq + 0) * B_PITCH + 2 * kp] = pk_bf16(rw0.x, rw1.x);
    *(unsigned*)&Btlds[buf][(nq + 1) * B_PITCH + 2 * kp] = pk_bf16(rw0.y, rw1.y);
    *(unsigned*)&Btlds[buf][(nq + 2) * B_PITCH + 2 * kp] = pk_bf16(rw0.z, rw1.z);
    *(unsigned*)&Btlds[buf][(nq + 3) * B_PITCH + 2 * kp] = pk_bf16(rw0.w, rw1.w);
  };

  v8f acc[2][2] = {};

  auto COMPUTE = [&](int buf) {
    const unsigned short* Ab = Alds[buf];
    const unsigned short* Bb = Btlds[buf];
    FragU a[2], b[2];
#pragma unroll
    for (int i = 0; i < 2; ++i) {
      const unsigned short* ap = Ab + (wm * 32 + i * 16 + l15) * A_PITCH;
      a[i].u[0] = *(const uint4*)(ap + hiA);
      a[i].u[1] = *(const uint4*)(ap + 16 + hiA);
    }
#pragma unroll
    for (int j = 0; j < 2; ++j) {
      const unsigned short* bp = Bb + (wn * 32 + j * 16 + l15) * B_PITCH + hiA;
      b[j].u[0] = *(const uint4*)(bp);
      b[j].u[1] = *(const uint4*)(bp + 16);
    }
#pragma unroll
    for (int i = 0; i < 2; ++i)
#pragma unroll
      for (int j = 0; j < 2; ++j)
        acc[i][j] = __builtin_amdgcn_wmma_f32_16x16x32_bf16(
            false, a[i].v, false, b[j].v, (short)0, acc[i][j], false, false);
  };

  const int nk = D / K_BLK;            // D=2048 -> 64 (even)

  GLOAD(0);
  SSTORE(0);
  __syncthreads();

  for (int k = 0; k < nk; k += 2) {
    // even step: compute buf0, stage k+1 into buf1
    if (k + 1 < nk) GLOAD((k + 1) * K_BLK);
    COMPUTE(0);
    if (k + 1 < nk) { SSTORE(1); __syncthreads(); }

    if (k + 1 < nk) {
      // odd step: compute buf1, stage k+2 into buf0
      if (k + 2 < nk) GLOAD((k + 2) * K_BLK);
      COMPUTE(1);
      if (k + 2 < nk) { SSTORE(0); __syncthreads(); }
    }
  }

  // epilogue: D layout -> global, fuse bias
#pragma unroll
  for (int j = 0; j < 2; ++j) {
    int c = n_base + wn * 32 + j * 16 + l15;
    float bb = bias[c];
#pragma unroll
    for (int i = 0; i < 2; ++i) {
      int rbase = m_base + wm * 32 + i * 16 + ((lane >> 4) << 3);
#pragma unroll
      for (int e = 0; e < 8; ++e)
        out[(size_t)(rbase + e) * H + c] = acc[i][j][e] + bb;
    }
  }
}

// ---------------------------------------------------------------------------
// q0 = gw0 @ Wq + bq
// ---------------------------------------------------------------------------
__global__ __launch_bounds__(256) void q0_kernel(
    const float* __restrict__ gw0, const float* __restrict__ Wq,
    const float* __restrict__ bq, float* __restrict__ q0, int D, int H) {
  int h = blockIdx.x * blockDim.x + threadIdx.x;
  if (h >= H) return;
  float s = bq[h];
  for (int d = 0; d < D; d += 4) {
    s += gw0[d]     * Wq[(size_t)d * H + h];
    s += gw0[d + 1] * Wq[(size_t)(d + 1) * H + h];
    s += gw0[d + 2] * Wq[(size_t)(d + 2) * H + h];
    s += gw0[d + 3] * Wq[(size_t)(d + 3) * H + h];
  }
  q0[h] = s;
}

// ---------------------------------------------------------------------------
// One wave32 per row: 6 dot products, then the full N_STEPS recurrence as
// scalar math (softmax weights sum to 1; bq folded into P_v/P_t).
// ---------------------------------------------------------------------------
__global__ __launch_bounds__(256) void attn_iter_kernel(
    const float* __restrict__ kv, const float* __restrict__ kt,
    const float* __restrict__ Pv, const float* __restrict__ Pt,
    const float* __restrict__ q0, float* __restrict__ out, int H) {
  const int lane = threadIdx.x & 31;
  const int row  = blockIdx.x * 8 + (threadIdx.x >> 5);
  const size_t base = (size_t)row * H;

  float cvv = 0.f, cvt_ = 0.f, ctv = 0.f, ctt = 0.f, cv0 = 0.f, ct0 = 0.f;
  for (int i4 = lane; i4 < (H >> 2); i4 += 32) {
    int idx = i4 << 2;
    float4 a  = *(const float4*)(kv + base + idx);
    float4 b  = *(const float4*)(kt + base + idx);
    float4 pv = *(const float4*)(Pv + base + idx);
    float4 pt = *(const float4*)(Pt + base + idx);
    float4 q  = *(const float4*)(q0 + idx);
    cvv  += a.x * pv.x + a.y * pv.y + a.z * pv.z + a.w * pv.w;
    cvt_ += a.x * pt.x + a.y * pt.y + a.z * pt.z + a.w * pt.w;
    ctv  += b.x * pv.x + b.y * pv.y + b.z * pv.z + b.w * pv.w;
    ctt  += b.x * pt.x + b.y * pt.y + b.z * pt.z + b.w * pt.w;
    cv0  += a.x * q.x  + a.y * q.y  + a.z * q.z  + a.w * q.w;
    ct0  += b.x * q.x  + b.y * q.y  + b.z * q.z  + b.w * q.w;
  }
#pragma unroll
  for (int off = 16; off > 0; off >>= 1) {
    cvv  += __shfl_xor(cvv,  off);
    cvt_ += __shfl_xor(cvt_, off);
    ctv  += __shfl_xor(ctv,  off);
    ctt  += __shfl_xor(ctt,  off);
    cv0  += __shfl_xor(cv0,  off);
    ct0  += __shfl_xor(ct0,  off);
  }
  if (lane == 0) {
    float d0 = cv0, d1 = ct0, a0 = 0.f, a1 = 0.f;
#pragma unroll
    for (int s = 0; s <= N_STEPS; ++s) {
      float m  = fmaxf(d0, d1);
      float e0 = __expf(d0 - m), e1 = __expf(d1 - m);
      float inv = 1.0f / (e0 + e1);
      a0 = e0 * inv; a1 = e1 * inv;
      if (s < N_STEPS) {
        d0 = a0 * cvv + a1 * cvt_;
        d1 = a0 * ctv + a1 * ctt;
      }
    }
    out[row * 2 + 0] = a0;
    out[row * 2 + 1] = a1;
  }
}

// ---------------------------------------------------------------------------
extern "C" void kernel_launch(void* const* d_in, const int* in_sizes, int n_in,
                              void* d_out, int out_size, void* d_ws, size_t ws_size,
                              hipStream_t stream) {
  const float* x_v  = (const float*)d_in[0];
  const float* x_t  = (const float*)d_in[1];
  const float* e_v  = (const float*)d_in[2];
  const float* e_t  = (const float*)d_in[3];
  const float* gw0  = (const float*)d_in[4];
  const float* Wq   = (const float*)d_in[5];
  const float* bq   = (const float*)d_in[6];
  const float* Wk_v = (const float*)d_in[7];
  const float* bk_v = (const float*)d_in[8];
  const float* Wk_t = (const float*)d_in[9];
  const float* bk_t = (const float*)d_in[10];

  const int D = in_sizes[4];
  const int H = in_sizes[6];
  const int B = in_sizes[0] / D;

  float* ws = (float*)d_ws;
  const size_t BH = (size_t)B * H;
  float* k_v = ws;
  float* k_t = ws + BH;
  float* P_v = ws + 2 * BH;
  float* P_t = ws + 3 * BH;
  float* q0  = ws + 4 * BH;

  dim3 ggrid(H / N_BLK, B / M_BLK, 4);
  gemm_bias_bf16<<<ggrid, 256, 0, stream>>>(x_v, x_t, e_v, e_t,
                                            Wk_v, Wk_t, Wq,
                                            bk_v, bk_t, bq,
                                            ws, B, D, H);
  q0_kernel<<<(H + 255) / 256, 256, 0, stream>>>(gw0, Wq, bq, q0, D, H);
  attn_iter_kernel<<<B / 8, 256, 0, stream>>>(k_v, k_t, P_v, P_t, q0,
                                              (float*)d_out, H);
}
